// DiagonalStripSSMGate_29076928593995
// MI455X (gfx1250) — matrix operand order repaired and checked by reading, not compiled, not run on hardware
//
#include <hip/hip_runtime.h>
#include <hip/hip_bf16.h>

// ---------------------------------------------------------------------------
// Types
// ---------------------------------------------------------------------------
typedef __attribute__((ext_vector_type(16))) _Float16 v16h;
typedef __attribute__((ext_vector_type(8)))  _Float16 v8h;
typedef __attribute__((ext_vector_type(8)))  float    v8f;
typedef __attribute__((ext_vector_type(4)))  float    f32x4;

union Frag16 { v16h h; f32x4 f[2]; };
union Pack8h { v8h h; f32x4 f; };

#define T_LEN   4096
#define D_LEN   256
#define C_LEN   64
#define TP_LEN  4112          // T + 16 halo rows (8 each side) in [t][c] layout
#define TILE_T  32

__device__ __forceinline__ float warp_red(float v) {
#pragma unroll
  for (int o = 16; o > 0; o >>= 1) v += __shfl_xor(v, o, 32);
  return v;
}

// gfx1250 async global->LDS copy (ASYNCcnt). LDS operand = addr[31:0] of the
// flat shared-aperture pointer (ISA 10.2).
__device__ __forceinline__ void async_b128(const void* g, const void* l) {
  unsigned loff = (unsigned)(unsigned long long)l;
  asm volatile("global_load_async_to_lds_b128 %0, %1, off"
               :: "v"(loff), "v"(g) : "memory");
}
__device__ __forceinline__ void async_wait0() {
  asm volatile("s_wait_asynccnt 0" ::: "memory");
}

// h buffers live in [b][t_padded][c] layout, fp16, 128B per time step
__device__ __forceinline__ size_t hidx(int b, int tpad, int c) {
  return ((size_t)b * TP_LEN + tpad) * C_LEN + c;
}

// ---------------------------------------------------------------------------
// Kernel 0: pack weights (fold w_in band symmetry, fp32 -> fp16 conv weights)
// WdH layout: [stage][tap][cout][cin], WpH: [stage][cout][cin]
// ---------------------------------------------------------------------------
__global__ void pack_weights_kernel(const float* __restrict__ w_in,
                                    const float* __restrict__ wd1,
                                    const float* __restrict__ wd2,
                                    const float* __restrict__ wd3,
                                    const float* __restrict__ wp1,
                                    const float* __restrict__ wp2,
                                    const float* __restrict__ wp3,
                                    float* __restrict__ wfold,
                                    _Float16* __restrict__ WdH,
                                    _Float16* __restrict__ WpH) {
  int tid = threadIdx.x;
  // fold w_in (C,17,1): strip rows +-a are identical
  for (int e = tid; e < 64 * 9; e += 256) {
    int c = e / 9, a = e % 9;
    float v = (a == 0) ? w_in[c * 17 + 8]
                       : (w_in[c * 17 + 8 - a] + w_in[c * 17 + 8 + a]);
    wfold[c * 9 + a] = v;
  }
  // dilated conv weights (C,C,3) -> [tap][cout][cin] fp16
  for (int e = tid; e < 3 * 3 * 64 * 64; e += 256) {
    int s = e / 12288, r = e % 12288;
    int j = r / 4096;  r %= 4096;
    int m = r / 64;    int k = r % 64;
    const float* src = (s == 0) ? wd1 : (s == 1) ? wd2 : wd3;
    WdH[e] = (_Float16)src[m * 192 + k * 3 + j];
  }
  // pointwise conv weights (C,C,1) -> [cout][cin] fp16
  for (int e = tid; e < 3 * 64 * 64; e += 256) {
    int s = e / 4096, r = e % 4096;
    int m = r / 64, k = r % 64;
    const float* src = (s == 0) ? wp1 : (s == 1) ? wp2 : wp3;
    WpH[e] = (_Float16)src[m * 64 + k];
  }
}

// ---------------------------------------------------------------------------
// Kernel 1: zero the 8-row halos (t in [0,8) and [TP-8,TP)) of both buffers
// ---------------------------------------------------------------------------
__global__ void zero_halo_kernel(_Float16* __restrict__ hA,
                                 _Float16* __restrict__ hB) {
  int idx = blockIdx.x * 256 + threadIdx.x;   // 2 bufs * 32 b * 16 t * 64 c
  _Float16* buf = (idx >= 32768) ? hB : hA;
  int rem = idx & 32767;
  int b  = rem >> 10;
  int r2 = rem & 1023;
  int tq = r2 >> 6;
  int c  = r2 & 63;
  int t  = (tq < 8) ? tq : (TP_LEN - 16 + tq);
  buf[hidx(b, t, c)] = (_Float16)0.0f;
}

// ---------------------------------------------------------------------------
// Kernel 2: banded self-similarity strip (9 folded bands) + fused 1x1 in-proj
// Raw feature rows async-staged to LDS; normalization folded into per-row
// scalars: dot(xn_t,xn_s) = sc_t * sc_s * dot(feat_t, feat_s),
// sc_t = m_t / (|m_t| * ||feat_t|| + eps)   (== reference exactly)
// ---------------------------------------------------------------------------
__global__ void strip_h0_kernel(const float* __restrict__ feat,
                                const float* __restrict__ mask,
                                const float* __restrict__ wfold,
                                const float* __restrict__ b_in,
                                _Float16* __restrict__ h0) {
  __shared__ float xmL[40 * 260];   // raw rows, stride 260 (16B aligned, low conflict)
  __shared__ float rowscale[40];
  __shared__ float bands[9 * 33];

  const int b  = blockIdx.y;
  const int t0 = blockIdx.x * TILE_T;
  const int tid = threadIdx.x;
  const int wid = tid >> 5, lane = tid & 31;

  // async stage 40 rows x 256 floats (16B chunks)
  for (int q = tid; q < 40 * 64; q += 256) {
    int r = q >> 6, chk = q & 63;
    int t = t0 + r;
    if (t < T_LEN) {
      const float* g = feat + ((size_t)b * T_LEN + t) * D_LEN + chk * 4;
      async_b128(g, &xmL[r * 260 + chk * 4]);
    }
  }
  async_wait0();
  __syncthreads();

  // per-row normalization scalars
  for (int r = wid; r < 40; r += 8) {
    int t = t0 + r;
    float ss = 0.0f;
    if (t < T_LEN) {
      const float* row = &xmL[r * 260 + lane * 8];
#pragma unroll
      for (int k = 0; k < 8; ++k) ss += row[k] * row[k];
    }
    ss = warp_red(ss);
    if (lane == 0) {
      float sc = 0.0f;
      if (t < T_LEN) {
        float mval = mask[(size_t)b * T_LEN + t];
        sc = mval / (fabsf(mval) * sqrtf(ss) + 1e-6f);
      }
      rowscale[r] = sc;
    }
  }
  __syncthreads();

  // 9 bands x 32 positions
  for (int idx = tid; idx < 9 * 32; idx += 256) {
    int a = idx >> 5, tl = idx & 31;
    int t = t0 + tl;
    float s = 0.0f;
    if (t + a < T_LEN) {
      const float* r0 = &xmL[tl * 260];
      const float* r1 = &xmL[(tl + a) * 260];
      float acc = 0.0f;
#pragma unroll 8
      for (int k = 0; k < D_LEN; ++k) acc += r0[k] * r1[k];
      acc *= rowscale[tl] * rowscale[tl + a];
      float mv = mask[(size_t)b * T_LEN + t] * mask[(size_t)b * T_LEN + t + a];
      s = (acc - ((a == 0) ? 0.1f : 0.0f) + 1.0f) * 0.5f * mv;
    }
    bands[a * 33 + tl] = s;
  }
  __syncthreads();

  // h0[t,c] = relu(b_in[c] + sum_a wfold[c,a] * band[a,t])  (fp16, [t][c])
  for (int e = tid; e < C_LEN * TILE_T; e += 256) {
    int c = e & 63, tl = e >> 6;
    float acc = b_in[c];
#pragma unroll
    for (int a = 0; a < 9; ++a) acc += wfold[c * 9 + a] * bands[a * 33 + tl];
    h0[hidx(b, 8 + t0 + tl, c)] = (_Float16)fmaxf(acc, 0.0f);
  }
}

// ---------------------------------------------------------------------------
// Kernel 3: fused stage: y = relu(conv1x1(Wp, relu(conv3_dil(Wd, h)+bd))+bp)
// WMMA f16 16x16x32. [t][c] layout => async tile staging is a straight copy
// already in the order B-fragments need; output is one b128 store per lane.
// ---------------------------------------------------------------------------
__global__ void stage_kernel(const _Float16* __restrict__ hin,
                             _Float16* __restrict__ hout,
                             const _Float16* __restrict__ WdHs,
                             const float* __restrict__ bd,
                             const _Float16* __restrict__ WpHs,
                             const float* __restrict__ bp,
                             int dil) {
  __shared__ _Float16 tileIn[40 * 72];   // [t_local][cin], stride 72 halves
  __shared__ _Float16 y1[32 * 72];       // [t_local][cout]

  const int b   = blockIdx.y;
  const int t0  = blockIdx.x * TILE_T;
  const int tid = threadIdx.x;
  const int wid = tid >> 5, lane = tid & 31;
  const int ntl = TILE_T + 2 * dil;

  if (tid == 0 && blockIdx.x + 1 < gridDim.x)  // hint next tile into cache
    __builtin_prefetch(hin + hidx(b, 8 + t0 + TILE_T, 0), 0, 1);

  // async stage input tile: ntl rows x 64 ch (128B = 8 x b128 per row)
  for (int q = tid; q < ntl * 8; q += 256) {
    int tl = q >> 3, chk = q & 7;
    const _Float16* g = hin + hidx(b, 8 + t0 - dil + tl, chk * 8);
    async_b128(g, &tileIn[tl * 72 + chk * 8]);
  }

  const int mt = wid >> 1;          // M tile 0..3  (out channels mt*16..)
  const int nt = wid & 1;           // N tile 0..1  (t positions nt*16..)
  const int mrow  = lane & 15;      // A-fragment: lane -> M row
  const int koffA = (lane >= 16) ? 8 : 0;
  const int nIdx  = lane & 15;      // B/C/D: lane -> N col
  const int koffB = (lane >= 16) ? 8 : 0;

  // preload A fragments (weights): 6 for conv3, 2 for conv1
  Frag16 Ad[6], Ap[2];
#pragma unroll
  for (int j = 0; j < 3; ++j)
#pragma unroll
    for (int kc = 0; kc < 2; ++kc) {
      const f32x4* p = (const f32x4*)(WdHs +
          ((size_t)(j * 64 + mt * 16 + mrow)) * 64 + kc * 32 + koffA);
      Ad[j * 2 + kc].f[0] = p[0];   // K run [koff, koff+8)
      Ad[j * 2 + kc].f[1] = p[2];   // K run [koff+16, koff+24)
    }
#pragma unroll
  for (int kc = 0; kc < 2; ++kc) {
    const f32x4* p = (const f32x4*)(WpHs +
        ((size_t)(mt * 16 + mrow)) * 64 + kc * 32 + koffA);
    Ap[kc].f[0] = p[0];
    Ap[kc].f[1] = p[2];
  }

  async_wait0();
  __syncthreads();

  // GEMM 1: dilated conv3, K = 3 taps x 64 ch = 6 WMMA steps
  v8f acc;
#pragma unroll
  for (int i = 0; i < 8; ++i) acc[i] = 0.0f;
#pragma unroll
  for (int j = 0; j < 3; ++j) {
    int tl = nt * 16 + nIdx + j * dil;
#pragma unroll
    for (int kc = 0; kc < 2; ++kc) {
      const f32x4* bp4 = (const f32x4*)(tileIn + tl * 72 + kc * 32 + koffB);
      Frag16 B;
      B.f[0] = bp4[0];
      B.f[1] = bp4[2];
      acc = __builtin_amdgcn_wmma_f32_16x16x32_f16(
          false, Ad[j * 2 + kc].h, false, B.h, (short)0, acc, false, false);
    }
  }

  // bias + relu -> y1 (LDS, [t][c], fp16)
  const int m0 = mt * 16 + ((lane >= 16) ? 8 : 0);
  {
    Pack8h pk;
#pragma unroll
    for (int v = 0; v < 8; ++v)
      pk.h[v] = (_Float16)fmaxf(acc[v] + bd[m0 + v], 0.0f);
    *(f32x4*)(y1 + (nt * 16 + nIdx) * 72 + m0) = pk.f;
  }
  __syncthreads();

  // GEMM 2: pointwise conv1, K = 64 = 2 WMMA steps
  v8f acc2;
#pragma unroll
  for (int i = 0; i < 8; ++i) acc2[i] = 0.0f;
#pragma unroll
  for (int kc = 0; kc < 2; ++kc) {
    const f32x4* bp4 = (const f32x4*)(y1 + (nt * 16 + nIdx) * 72 + kc * 32 + koffB);
    Frag16 B;
    B.f[0] = bp4[0];
    B.f[1] = bp4[2];
    acc2 = __builtin_amdgcn_wmma_f32_16x16x32_f16(
        false, Ap[kc].h, false, B.h, (short)0, acc2, false, false);
  }

  // bias + relu -> one aligned 16B global store per lane ([t][c] layout)
  {
    Pack8h pk;
#pragma unroll
    for (int v = 0; v < 8; ++v)
      pk.h[v] = (_Float16)fmaxf(acc2[v] + bp[m0 + v], 0.0f);
    *(f32x4*)(hout + hidx(b, 8 + t0 + nt * 16 + nIdx, m0)) = pk.f;
  }
}

// ---------------------------------------------------------------------------
// Kernel 4: masked mean pool over T -> z[b, zbase + c]  (coalesced in [t][c])
// ---------------------------------------------------------------------------
__global__ void pool_kernel(const _Float16* __restrict__ h,
                            const float* __restrict__ mask,
                            float* __restrict__ zbuf, int zbase) {
  __shared__ float pacc[4 * 64];
  __shared__ float pm[4];
  const int b = blockIdx.x;
  const int tid = threadIdx.x;
  const int c = tid & 63, slot = tid >> 6;       // 4 slots x 64 channels
  float acc = 0.0f, macc = 0.0f;
  for (int t = slot; t < T_LEN; t += 4) {
    float mv = mask[(size_t)b * T_LEN + t];
    acc += (float)h[hidx(b, 8 + t, c)] * mv;
    macc += mv;
  }
  pacc[slot * 64 + c] = acc;
  if (c == 0) pm[slot] = macc;
  __syncthreads();
  if (tid < 64) {
    float tot = pacc[tid] + pacc[64 + tid] + pacc[128 + tid] + pacc[192 + tid];
    float ms = pm[0] + pm[1] + pm[2] + pm[3];
    zbuf[b * 128 + zbase + tid] = tot / (ms + 1e-6f);
  }
}

// ---------------------------------------------------------------------------
// Kernel 5: LayerNorm + two gate heads (sigmoid / tanh). One block.
// ---------------------------------------------------------------------------
__global__ void head_kernel(const float* __restrict__ zbuf,
                            const float* __restrict__ ln_g,
                            const float* __restrict__ ln_b,
                            const float* __restrict__ mw1, const float* __restrict__ mb1,
                            const float* __restrict__ mw2, const float* __restrict__ mb2,
                            const float* __restrict__ fw1, const float* __restrict__ fb1,
                            const float* __restrict__ fw2, const float* __restrict__ fb2,
                            float* __restrict__ out) {
  __shared__ float fnS[32 * 132];
  const int tid = threadIdx.x, wid = tid >> 5, lane = tid & 31;

  // LayerNorm per row (128 features; lane holds cols lane+32j)
  for (int b = wid; b < 32; b += 8) {
    float f[4];
#pragma unroll
    for (int j = 0; j < 4; ++j) f[j] = zbuf[b * 128 + lane + 32 * j];
    float s = f[0] + f[1] + f[2] + f[3];
    float mu = warp_red(s) * (1.0f / 128.0f);
    float vs = 0.0f;
#pragma unroll
    for (int j = 0; j < 4; ++j) { float d = f[j] - mu; vs += d * d; }
    float var = warp_red(vs) * (1.0f / 128.0f);
    float rs = rsqrtf(var + 1e-5f);
#pragma unroll
    for (int j = 0; j < 4; ++j) {
      int col = lane + 32 * j;
      fnS[b * 132 + col] = (f[j] - mu) * rs * ln_g[col] + ln_b[col];
    }
  }
  __syncthreads();

  // MLP heads: fn(128) -> relu 64 -> 1
  for (int b = wid; b < 32; b += 8) {
    int c0 = lane, c1 = lane + 32;
    float am0 = mb1[c0], am1 = mb1[c1];
    float af0 = fb1[c0], af1 = fb1[c1];
    for (int k = 0; k < 128; ++k) {
      float fv = fnS[b * 132 + k];
      am0 += fv * mw1[k * 64 + c0];
      am1 += fv * mw1[k * 64 + c1];
      af0 += fv * fw1[k * 64 + c0];
      af1 += fv * fw1[k * 64 + c1];
    }
    float pm = fmaxf(am0, 0.0f) * mw2[c0] + fmaxf(am1, 0.0f) * mw2[c1];
    float pf = fmaxf(af0, 0.0f) * fw2[c0] + fmaxf(af1, 0.0f) * fw2[c1];
    pm = warp_red(pm);
    pf = warp_red(pf);
    if (lane == 0) {
      out[b]      = 1.0f / (1.0f + expf(-(pm + mb2[0])));   // sigmoid gate
      out[32 + b] = tanhf(pf + fb2[0]);                     // tanh gate
    }
  }
}

// ---------------------------------------------------------------------------
// Host side
// ---------------------------------------------------------------------------
extern "C" void kernel_launch(void* const* d_in, const int* in_sizes, int n_in,
                              void* d_out, int out_size, void* d_ws, size_t ws_size,
                              hipStream_t stream) {
  const float* video_feat = (const float*)d_in[0];
  const float* audio_feat = (const float*)d_in[1];
  const float* video_mask = (const float*)d_in[2];
  const float* audio_mask = (const float*)d_in[3];
  const float* w_in = (const float*)d_in[4];
  const float* b_in = (const float*)d_in[5];
  const float* w_d[3] = {(const float*)d_in[6], (const float*)d_in[10], (const float*)d_in[14]};
  const float* b_d[3] = {(const float*)d_in[7], (const float*)d_in[11], (const float*)d_in[15]};
  const float* w_p[3] = {(const float*)d_in[8], (const float*)d_in[12], (const float*)d_in[16]};
  const float* b_p[3] = {(const float*)d_in[9], (const float*)d_in[13], (const float*)d_in[17]};
  const float* ln_g = (const float*)d_in[18];
  const float* ln_b = (const float*)d_in[19];
  const float* mw1 = (const float*)d_in[20];
  const float* mb1 = (const float*)d_in[21];
  const float* mw2 = (const float*)d_in[22];
  const float* mb2 = (const float*)d_in[23];
  const float* fw1 = (const float*)d_in[24];
  const float* fb1 = (const float*)d_in[25];
  const float* fw2 = (const float*)d_in[26];
  const float* fb2 = (const float*)d_in[27];
  float* out = (float*)d_out;

  // workspace layout (16B-aligned offsets)
  char* ws = (char*)d_ws;
  const size_t OFF_WFOLD = 0;                              // 576 f32
  const size_t OFF_WDH   = 2304;                           // 3*3*64*64 f16
  const size_t OFF_WPH   = OFF_WDH + 73728;                // 3*64*64 f16
  const size_t OFF_Z     = OFF_WPH + 24576;                // 32*128 f32
  const size_t OFF_HA    = OFF_Z + 16384;
  const size_t HBYTES    = (size_t)32 * TP_LEN * C_LEN * sizeof(_Float16);
  const size_t OFF_HB    = OFF_HA + HBYTES;

  float*    wfold = (float*)(ws + OFF_WFOLD);
  _Float16* WdH   = (_Float16*)(ws + OFF_WDH);
  _Float16* WpH   = (_Float16*)(ws + OFF_WPH);
  float*    zbuf  = (float*)(ws + OFF_Z);
  _Float16* hA    = (_Float16*)(ws + OFF_HA);
  _Float16* hB    = (_Float16*)(ws + OFF_HB);

  pack_weights_kernel<<<1, 256, 0, stream>>>(w_in, w_d[0], w_d[1], w_d[2],
                                             w_p[0], w_p[1], w_p[2],
                                             wfold, WdH, WpH);
  zero_halo_kernel<<<256, 256, 0, stream>>>(hA, hB);

  const int dils[3] = {1, 2, 4};
  dim3 gridT(T_LEN / TILE_T, 32);

  // audio -> z cols [0,64), video -> z cols [64,128)
  for (int mod = 0; mod < 2; ++mod) {
    const float* feat = (mod == 0) ? audio_feat : video_feat;
    const float* msk  = (mod == 0) ? audio_mask : video_mask;
    int zbase = (mod == 0) ? 0 : 64;

    strip_h0_kernel<<<gridT, 256, 0, stream>>>(feat, msk, wfold, b_in, hA);

    _Float16* src = hA;
    _Float16* dst = hB;
    for (int s = 0; s < 3; ++s) {
      stage_kernel<<<gridT, 256, 0, stream>>>(src, dst,
                                              WdH + (size_t)s * 3 * 64 * 64, b_d[s],
                                              WpH + (size_t)s * 64 * 64, b_p[s],
                                              dils[s]);
      _Float16* tmp = src; src = dst; dst = tmp;
    }
    pool_kernel<<<32, 256, 0, stream>>>(src, msk, zbuf, zbase);
  }

  head_kernel<<<1, 256, 0, stream>>>(zbuf, ln_g, ln_b,
                                     mw1, mb1, mw2, mb2,
                                     fw1, fb1, fw2, fb2, out);
}
